// ConCare_58969900974527
// MI455X (gfx1250) — matrix operand order repaired and checked by reading, not compiled, not run on hardware
//
#include <hip/hip_runtime.h>

// ---------------- types & helpers ----------------
typedef __attribute__((ext_vector_type(16))) __bf16 v16bf;
typedef __attribute__((ext_vector_type(8)))  float  v8f;

union Frag16 { v16bf v; unsigned short u[16]; };

__device__ __forceinline__ unsigned short f2bf(float f) {
    unsigned int u = __float_as_uint(f);
    u += 0x7fffu + ((u >> 16) & 1u);      // round-to-nearest-even
    return (unsigned short)(u >> 16);
}
__device__ __forceinline__ float bf2f(unsigned short h) {
    return __uint_as_float(((unsigned int)h) << 16);
}
__device__ __forceinline__ float sigmoidf_(float x) { return 1.f / (1.f + expf(-x)); }

// A fragment: 16x32 bf16, row-major source, leading dim ld (elements).
// Lane 0-15: M=lane, K = 0..15; lane 16-31: M=lane-16, K = 16..31.
__device__ __forceinline__ v16bf load_a_frag(const unsigned short* s, int ld, int lane) {
    Frag16 f;
    int m  = lane & 15;
    int kb = (lane >> 4) << 4;          // 0 or 16
    const unsigned short* row = s + m * ld;
#pragma unroll
    for (int i = 0; i < 16; i++) {
        f.u[i] = row[kb + i];
    }
    return f.v;
}
// B fragment: 32x16 bf16, row-major [K][N] source, leading dim ld.
// Lane 0-15: N=lane, K=0..15; lane 16-31: N=lane-16, K=16..31.
__device__ __forceinline__ v16bf load_b_frag(const unsigned short* s, int ld, int lane) {
    Frag16 f;
    int n    = lane & 15;
    int koff = (lane >> 4) << 4;        // 0 or 16
#pragma unroll
    for (int i = 0; i < 16; i++) f.u[i] = s[(koff + i) * ld + n];
    return f.v;
}
__device__ __forceinline__ v8f wmma_bf16(v16bf a, v16bf b, v8f c) {
    return __builtin_amdgcn_wmma_f32_16x16x32_bf16(false, a, false, b, (short)0, c, false, false);
}
// C/D element (r, lane): row = r + (lane>=16)*8, col = lane&15

// ---------------- dims ----------------
#define BZ   64
#define TZ   128
#define FZ   64
#define HZ   128
#define AZ   8
#define NHD  4
#define DKH  32
#define DFFZ 512
#define DDZ  12
#define NZ   65     // F + 1 tokens

// ---------------- conversion kernels ----------------
__global__ void conv_bf_kernel(const float* __restrict__ s, unsigned short* __restrict__ d, int n) {
    for (int i = blockIdx.x * blockDim.x + threadIdx.x; i < n; i += gridDim.x * blockDim.x)
        d[i] = f2bf(s[i]);
}
// att_wx [F][H][8] -> padded bf16 [F][H][16] (cols 8..15 zero)
__global__ void conv_wx_kernel(const float* __restrict__ wx, unsigned short* __restrict__ d) {
    int n = FZ * HZ * 16;
    for (int i = blockIdx.x * blockDim.x + threadIdx.x; i < n; i += gridDim.x * blockDim.x) {
        int a = i & 15;
        int rest = i >> 4;              // f*H + h
        d[i] = (a < AZ) ? f2bf(wx[rest * AZ + a]) : (unsigned short)0;
    }
}

// ---------------- 1) per-feature GRU (weight-stationary WMMA) ----------------
__global__ void __launch_bounds__(256) gru_kernel(const float* __restrict__ x,
                                                  const float* __restrict__ wih,
                                                  const float* __restrict__ bih,
                                                  const float* __restrict__ bhh,
                                                  const unsigned short* __restrict__ whh_bf,
                                                  unsigned short* __restrict__ Hseq) {
    int f    = blockIdx.x;
    int tid  = threadIdx.x;
    int wave = tid >> 5, lane = tid & 31;

    __shared__ float          h_f32[BZ * HZ];     // 32 KB
    __shared__ unsigned short h_bf [BZ * HZ];     // 16 KB
    __shared__ float          xt[BZ];
    __shared__ float          wih_s[3 * HZ], bih_s[3 * HZ], bhh_s[3 * HZ];

    for (int i = tid; i < BZ * HZ; i += 256) { h_f32[i] = 0.f; h_bf[i] = 0; }
    for (int i = tid; i < 3 * HZ; i += 256) {
        wih_s[i] = wih[f * 3 * HZ + i];
        bih_s[i] = bih[f * 3 * HZ + i];
        bhh_s[i] = bhh[f * 3 * HZ + i];
    }

    // Weight-stationary B fragments: this wave owns hidden cols [jb, jb+16) of all 3 gates.
    const unsigned short* whhf = whh_bf + f * HZ * (3 * HZ);
    int jb = wave * 16;
    v16bf Bf[3][4];
#pragma unroll
    for (int g = 0; g < 3; g++)
#pragma unroll
        for (int kt = 0; kt < 4; kt++)
            Bf[g][kt] = load_b_frag(whhf + kt * 32 * (3 * HZ) + g * HZ + jb, 3 * HZ, lane);
    __syncthreads();

    int j  = jb + (lane & 15);
    float wr = wih_s[j], wz = wih_s[HZ + j], wn = wih_s[2 * HZ + j];
    float br = bih_s[j], bz = bih_s[HZ + j], bn = bih_s[2 * HZ + j];
    float hr = bhh_s[j], hz = bhh_s[HZ + j], hn = bhh_s[2 * HZ + j];

    for (int t = 0; t < TZ; t++) {
        if (tid < BZ) xt[tid] = x[(tid * TZ + t) * FZ + f];
        __syncthreads();

        v8f zero = {};
        v8f acc[3][4];
#pragma unroll
        for (int g = 0; g < 3; g++)
#pragma unroll
            for (int m = 0; m < 4; m++) acc[g][m] = zero;

#pragma unroll
        for (int kt = 0; kt < 4; kt++) {
#pragma unroll
            for (int m = 0; m < 4; m++) {
                v16bf a = load_a_frag(h_bf + m * 16 * HZ + kt * 32, HZ, lane);
                acc[0][m] = wmma_bf16(a, Bf[0][kt], acc[0][m]);
                acc[1][m] = wmma_bf16(a, Bf[1][kt], acc[1][m]);
                acc[2][m] = wmma_bf16(a, Bf[2][kt], acc[2][m]);
            }
        }
        __syncthreads();   // all reads of h_bf complete before updates

#pragma unroll
        for (int m = 0; m < 4; m++) {
#pragma unroll
            for (int r = 0; r < 8; r++) {
                int b = m * 16 + r + ((lane >> 4) << 3);
                float xv  = xt[b];
                float ghr = acc[0][m][r] + hr;
                float ghz = acc[1][m][r] + hz;
                float ghn = acc[2][m][r] + hn;
                float rg = sigmoidf_(xv * wr + br + ghr);
                float zg = sigmoidf_(xv * wz + bz + ghz);
                float ng = tanhf(xv * wn + bn + rg * ghn);
                float hprev = h_f32[b * HZ + j];
                float hnew  = (1.f - zg) * ng + zg * hprev;
                h_f32[b * HZ + j] = hnew;
                unsigned short hb = f2bf(hnew);
                h_bf[b * HZ + j]  = hb;
                Hseq[(((size_t)f * BZ + b) * TZ + t) * HZ + j] = hb;
            }
        }
        __syncthreads();
    }
}

// ---------------- 2) per-feature time-decay attention ----------------
__global__ void __launch_bounds__(256) time_attn_kernel(const unsigned short* __restrict__ Hseq,
                                                        const unsigned short* __restrict__ wx_bf,
                                                        const float* __restrict__ att_wt,
                                                        const float* __restrict__ att_rate,
                                                        float* __restrict__ feats) {
    int f = blockIdx.x, b = blockIdx.y;
    int tid = threadIdx.x, wave = tid >> 5, lane = tid & 31;
    const unsigned short* Hfb = Hseq + ((size_t)f * BZ + b) * TZ * HZ;
    const unsigned short* wx  = wx_bf + f * HZ * 16;

    __shared__ float ks[TZ * AZ];
    __shared__ float qv[AZ];
    __shared__ float es[TZ];
    __shared__ float als[TZ];

    // k = Hseq_fb (128x128) @ att_wx (128x16 padded) -> take cols 0..7
    {
        int mt = wave;
        v8f acc = {};
#pragma unroll
        for (int kt = 0; kt < 4; kt++) {
            v16bf a  = load_a_frag(Hfb + mt * 16 * HZ + kt * 32, HZ, lane);
            v16bf bb = load_b_frag(wx + kt * 32 * 16, 16, lane);
            acc = wmma_bf16(a, bb, acc);
        }
        int col = lane & 15;
        if (col < AZ) {
#pragma unroll
            for (int r = 0; r < 8; r++) {
                int t = mt * 16 + r + ((lane >> 4) << 3);
                ks[t * AZ + col] = acc[r];
            }
        }
    }
    // q = h_{T-1} @ att_wt  (tiny)
    if (tid < AZ) {
        float s = 0.f;
        for (int h = 0; h < HZ; h++)
            s += bf2f(Hfb[(TZ - 1) * HZ + h]) * att_wt[((size_t)f * HZ + h) * AZ + tid];
        qv[tid] = s;
    }
    __syncthreads();

    if (tid < TZ) {
        float dot = 0.f;
#pragma unroll
        for (int a = 0; a < AZ; a++) dot += ks[tid * AZ + a] * qv[a];
        float sig   = sigmoidf_(dot);
        float btime = (float)(TZ - tid);
        float rate  = sigmoidf_(att_rate[f]);
        float denom = rate * logf(2.72f + (1.f - sig)) * btime;
        float e = sig / denom;
        es[tid] = fmaxf(e, 0.f);
    }
    __syncthreads();
    if (tid < TZ) {
        float mx = -1e30f;
        for (int t = 0; t < TZ; t++) mx = fmaxf(mx, es[t]);
        float sum = 0.f;
        for (int t = 0; t < TZ; t++) sum += expf(es[t] - mx);
        als[tid] = expf(es[tid] - mx) / sum;
    }
    __syncthreads();
    if (tid < HZ) {
        float o = 0.f;
        for (int t = 0; t < TZ; t++) o += als[t] * bf2f(Hfb[t * HZ + tid]);
        feats[((size_t)b * NZ + f) * HZ + tid] = o;   // token n=f
    }
}

// ---------------- 3) demographics token ----------------
__global__ void demo_proj_kernel(const float* __restrict__ demo,
                                 const float* __restrict__ w,
                                 const float* __restrict__ bias,
                                 float* __restrict__ feats) {
    int idx = blockIdx.x * blockDim.x + threadIdx.x;
    if (idx >= BZ * HZ) return;
    int b = idx / HZ, h = idx % HZ;
    float s = bias[h];
#pragma unroll
    for (int d = 0; d < DDZ; d++) s += demo[b * DDZ + d] * w[d * HZ + h];
    feats[((size_t)b * NZ + (NZ - 1)) * HZ + h] = tanhf(s);
}

// ---------------- generic bf16-WMMA GEMM: C = act(A@B + bias [+ residual]) ----------------
__global__ void __launch_bounds__(256) gemm_bf16_kernel(const float* __restrict__ A, int lda,
                                                        const unsigned short* __restrict__ Bw,
                                                        const float* __restrict__ bias,
                                                        const float* __restrict__ residual,
                                                        float* __restrict__ C, int ldc,
                                                        int M, int N, int K, int act) {
    __shared__ unsigned short As[64 * 32];
    __shared__ unsigned short Bs[32 * 64];
    int tid = threadIdx.x, wave = tid >> 5, lane = tid & 31;
    int m0 = blockIdx.y * 64, n0 = blockIdx.x * 64;
    int mt = wave >> 1, ntb = (wave & 1) * 2;
    v8f zero = {};
    v8f acc0 = zero, acc1 = zero;

    for (int k0 = 0; k0 < K; k0 += 32) {
        for (int idx = tid; idx < 64 * 32; idx += 256) {
            int r = idx >> 5, c = idx & 31;
            int gr = m0 + r, gc = k0 + c;
            float v = (gr < M && gc < K) ? A[(size_t)gr * lda + gc] : 0.f;
            As[idx] = f2bf(v);
        }
        for (int idx = tid; idx < 32 * 64; idx += 256) {
            int r = idx >> 6, c = idx & 63;
            int gr = k0 + r, gc = n0 + c;
            Bs[idx] = (gr < K && gc < N) ? Bw[(size_t)gr * N + gc] : (unsigned short)0;
        }
        __syncthreads();
        v16bf a  = load_a_frag(As + mt * 16 * 32, 32, lane);
        v16bf b0 = load_b_frag(Bs + ntb * 16, 64, lane);
        v16bf b1 = load_b_frag(Bs + (ntb + 1) * 16, 64, lane);
        acc0 = wmma_bf16(a, b0, acc0);
        acc1 = wmma_bf16(a, b1, acc1);
        __syncthreads();
    }
#pragma unroll
    for (int half = 0; half < 2; half++) {
        v8f acc = half ? acc1 : acc0;
        int col = n0 + (ntb + half) * 16 + (lane & 15);
#pragma unroll
        for (int r = 0; r < 8; r++) {
            int row = m0 + mt * 16 + r + ((lane >> 4) << 3);
            if (row < M && col < N) {
                float v = acc[r];
                if (bias)     v += bias[col];
                if (residual) v += residual[(size_t)row * ldc + col];
                if (act == 1) v = fmaxf(v, 0.f);
                else if (act == 2) v = tanhf(v);
                C[(size_t)row * ldc + col] = v;
            }
        }
    }
}

// ---------------- 4) MHA scores/softmax/context (N=65, dk=32, 4 heads) ----------------
__global__ void mha_attn_kernel(const float* __restrict__ Q, const float* __restrict__ K,
                                const float* __restrict__ V, float* __restrict__ Xh) {
    int b = blockIdx.x, head = blockIdx.y;
    __shared__ float Ks[NZ * DKH];
    __shared__ float Vs[NZ * DKH];
    int tid = threadIdx.x;  // 128
    for (int idx = tid; idx < NZ * DKH; idx += blockDim.x) {
        int n = idx / DKH, d = idx % DKH;
        Ks[idx] = K[((size_t)b * NZ + n) * HZ + head * DKH + d];
        Vs[idx] = V[((size_t)b * NZ + n) * HZ + head * DKH + d];
    }
    __syncthreads();
    if (tid < NZ) {
        float q[DKH];
#pragma unroll
        for (int d = 0; d < DKH; d++) q[d] = Q[((size_t)b * NZ + tid) * HZ + head * DKH + d];
        float sc[NZ];
        float mx = -1e30f;
        for (int m = 0; m < NZ; m++) {
            float s = 0.f;
#pragma unroll
            for (int d = 0; d < DKH; d++) s += q[d] * Ks[m * DKH + d];
            s *= 0.17677669529663689f;   // 1/sqrt(32)
            sc[m] = s; mx = fmaxf(mx, s);
        }
        float sum = 0.f;
        for (int m = 0; m < NZ; m++) { sc[m] = expf(sc[m] - mx); sum += sc[m]; }
        float inv = 1.f / sum;
        for (int d = 0; d < DKH; d++) {
            float o = 0.f;
            for (int m = 0; m < NZ; m++) o += sc[m] * Vs[m * DKH + d];
            Xh[((size_t)b * NZ + tid) * HZ + head * DKH + d] = o * inv;
        }
    }
}

// ---------------- elementwise add ----------------
__global__ void add_kernel(const float* __restrict__ a, const float* __restrict__ b,
                           float* __restrict__ c, int n) {
    for (int i = blockIdx.x * blockDim.x + threadIdx.x; i < n; i += gridDim.x * blockDim.x)
        c[i] = a[i] + b[i];
}

// ---------------- DeCov ----------------
__global__ void decov_center_kernel(const float* __restrict__ mo,
                                    unsigned short* __restrict__ c_bf,
                                    unsigned short* __restrict__ cT_bf) {
    int idx = blockIdx.x * blockDim.x + threadIdx.x;
    if (idx >= NZ * HZ) return;
    int n = idx / HZ, d = idx % HZ;
    float s = 0.f;
    for (int b = 0; b < BZ; b++) s += mo[((size_t)b * NZ + n) * HZ + d];
    float mean = s * (1.f / BZ);
    for (int b = 0; b < BZ; b++) {
        float cv = mo[((size_t)b * NZ + n) * HZ + d] - mean;
        unsigned short h = f2bf(cv);
        c_bf [((size_t)n * BZ + b) * HZ + d] = h;   // per n: [B][D]
        cT_bf[((size_t)n * HZ + d) * BZ + b] = h;   // per n: [D][B]
    }
}

__global__ void __launch_bounds__(256) decov_cov_kernel(const unsigned short* __restrict__ c_bf,
                                                        const unsigned short* __restrict__ cT_bf,
                                                        float* __restrict__ accum) {
    int n = blockIdx.x;
    int tid = threadIdx.x, wave = tid >> 5, lane = tid & 31;
    const unsigned short* cn  = c_bf  + (size_t)n * BZ * HZ;   // [64][128]
    const unsigned short* ctn = cT_bf + (size_t)n * HZ * BZ;   // [128][64]
    int mt = wave;
    v8f zero = {};
    v8f acc[8];
#pragma unroll
    for (int i = 0; i < 8; i++) acc[i] = zero;
#pragma unroll
    for (int kt = 0; kt < 2; kt++) {
        v16bf a = load_a_frag(ctn + mt * 16 * BZ + kt * 32, BZ, lane);
#pragma unroll
        for (int nt = 0; nt < 8; nt++) {
            v16bf bb = load_b_frag(cn + kt * 32 * HZ + nt * 16, HZ, lane);
            acc[nt] = wmma_bf16(a, bb, acc[nt]);
        }
    }
    const float invb = 1.f / (float)(BZ - 1);
    float ss = 0.f, sd = 0.f;
#pragma unroll
    for (int nt = 0; nt < 8; nt++)
#pragma unroll
        for (int r = 0; r < 8; r++) {
            int row = mt * 16 + r + ((lane >> 4) << 3);
            int col = nt * 16 + (lane & 15);
            float g = acc[nt][r] * invb;
            float g2 = g * g;
            ss += g2;
            if (row == col) sd += g2;
        }
    __shared__ float red[256], red2[256];
    red[tid] = ss; red2[tid] = sd;
    __syncthreads();
    for (int s = 128; s > 0; s >>= 1) {
        if (tid < s) { red[tid] += red[tid + s]; red2[tid] += red2[tid + s]; }
        __syncthreads();
    }
    if (tid == 0) { atomicAdd(&accum[0], red[0]); atomicAdd(&accum[1], red2[0]); }
}

__global__ void zero2_kernel(float* a) { a[0] = 0.f; a[1] = 0.f; }
__global__ void decov_write_kernel(const float* __restrict__ accum, float* __restrict__ out) {
    out[BZ] = 0.5f * (accum[0] - accum[1]);
}

// ---------------- final QKV pooling + head ----------------
__global__ void fa_pool_kernel(const float* __restrict__ qf, const float* __restrict__ kf,
                               const float* __restrict__ vf, float* __restrict__ pooled) {
    int b = blockIdx.x, tid = threadIdx.x;  // 128
    __shared__ float ef[NZ];
    __shared__ float al[NZ];
    if (tid < NZ) {
        float s = 0.f;
        const float* kr = kf + ((size_t)b * NZ + tid) * HZ;
        const float* qr = qf + (size_t)b * HZ;
        for (int a = 0; a < HZ; a++) s += kr[a] * qr[a];
        ef[tid] = s;
    }
    __syncthreads();
    float mx = -1e30f;
    for (int n = 0; n < NZ; n++) mx = fmaxf(mx, ef[n]);
    float sum = 0.f;
    for (int n = 0; n < NZ; n++) sum += expf(ef[n] - mx);
    if (tid < NZ) al[tid] = expf(ef[tid] - mx) / sum;
    __syncthreads();
    if (tid < HZ) {
        float o = 0.f;
        for (int n = 0; n < NZ; n++) o += al[n] * vf[((size_t)b * NZ + n) * HZ + tid];
        pooled[(size_t)b * HZ + tid] = o;
    }
}

__global__ void out_final_kernel(const float* __restrict__ h0, const float* __restrict__ w,
                                 const float* __restrict__ bias, float* __restrict__ out) {
    int b = blockIdx.x * blockDim.x + threadIdx.x;
    if (b >= BZ) return;
    float s = bias[0];
    for (int k = 0; k < HZ; k++) s += h0[b * HZ + k] * w[k];
    out[b] = 1.f / (1.f + expf(-s));
}

// ---------------- host launcher ----------------
extern "C" void kernel_launch(void* const* d_in, const int* in_sizes, int n_in,
                              void* d_out, int out_size, void* d_ws, size_t ws_size,
                              hipStream_t stream) {
    const float* x       = (const float*)d_in[0];
    const float* demo    = (const float*)d_in[1];
    const float* gru_wih = (const float*)d_in[2];
    const float* gru_bih = (const float*)d_in[3];
    const float* gru_whh = (const float*)d_in[4];
    const float* gru_bhh = (const float*)d_in[5];
    const float* att_wt  = (const float*)d_in[6];
    const float* att_wx  = (const float*)d_in[7];
    const float* att_rate= (const float*)d_in[8];
    const float* demo_w  = (const float*)d_in[9];
    const float* demo_b  = (const float*)d_in[10];
    const float* mha_wq  = (const float*)d_in[11];
    const float* mha_bq  = (const float*)d_in[12];
    const float* mha_wk  = (const float*)d_in[13];
    const float* mha_bk  = (const float*)d_in[14];
    const float* mha_wv  = (const float*)d_in[15];
    const float* mha_bv  = (const float*)d_in[16];
    const float* mha_wo  = (const float*)d_in[17];
    const float* mha_bo  = (const float*)d_in[18];
    const float* ff_w1   = (const float*)d_in[19];
    const float* ff_b1   = (const float*)d_in[20];
    const float* ff_w2   = (const float*)d_in[21];
    const float* ff_b2   = (const float*)d_in[22];
    const float* fa_wq   = (const float*)d_in[23];
    const float* fa_bq   = (const float*)d_in[24];
    const float* fa_wk   = (const float*)d_in[25];
    const float* fa_bk   = (const float*)d_in[26];
    const float* fa_wv   = (const float*)d_in[27];
    const float* fa_bv   = (const float*)d_in[28];
    const float* out0_w  = (const float*)d_in[29];
    const float* out0_b  = (const float*)d_in[30];
    const float* out1_w  = (const float*)d_in[31];
    const float* out1_b  = (const float*)d_in[32];
    float* dout = (float*)d_out;

    char* ws = (char*)d_ws;
    size_t off = 0;
    auto alloc = [&](size_t bytes) -> char* {
        char* p = ws + off;
        off += (bytes + 255) & ~(size_t)255;
        return p;
    };

    unsigned short* whh_bf  = (unsigned short*)alloc((size_t)FZ * HZ * 3 * HZ * 2);
    unsigned short* wx_bf   = (unsigned short*)alloc((size_t)FZ * HZ * 16 * 2);
    unsigned short* wq_bf   = (unsigned short*)alloc(HZ * HZ * 2);
    unsigned short* wk_bf   = (unsigned short*)alloc(HZ * HZ * 2);
    unsigned short* wv_bf   = (unsigned short*)alloc(HZ * HZ * 2);
    unsigned short* wo_bf   = (unsigned short*)alloc(HZ * HZ * 2);
    unsigned short* ffw1_bf = (unsigned short*)alloc(HZ * DFFZ * 2);
    unsigned short* ffw2_bf = (unsigned short*)alloc(DFFZ * HZ * 2);
    unsigned short* faq_bf  = (unsigned short*)alloc(HZ * HZ * 2);
    unsigned short* fak_bf  = (unsigned short*)alloc(HZ * HZ * 2);
    unsigned short* fav_bf  = (unsigned short*)alloc(HZ * HZ * 2);
    unsigned short* o0_bf   = (unsigned short*)alloc(HZ * HZ * 2);
    unsigned short* Hseq_bf = (unsigned short*)alloc((size_t)FZ * BZ * TZ * HZ * 2);
    float* feats  = (float*)alloc((size_t)BZ * NZ * HZ * 4);
    float* Qb     = (float*)alloc((size_t)BZ * NZ * HZ * 4);
    float* Kb     = (float*)alloc((size_t)BZ * NZ * HZ * 4);
    float* Vb     = (float*)alloc((size_t)BZ * NZ * HZ * 4);
    float* Xh     = (float*)alloc((size_t)BZ * NZ * HZ * 4);
    float* mha_o  = (float*)alloc((size_t)BZ * NZ * HZ * 4);
    float* ctx    = (float*)alloc((size_t)BZ * NZ * HZ * 4);
    float* ffh    = (float*)alloc((size_t)BZ * NZ * DFFZ * 4);
    float* ctx2   = (float*)alloc((size_t)BZ * NZ * HZ * 4);
    unsigned short* c_bf  = (unsigned short*)alloc((size_t)NZ * BZ * HZ * 2);
    unsigned short* cT_bf = (unsigned short*)alloc((size_t)NZ * HZ * BZ * 2);
    float* qf     = (float*)alloc(BZ * HZ * 4);
    float* kf     = (float*)alloc((size_t)BZ * NZ * HZ * 4);
    float* vf     = (float*)alloc((size_t)BZ * NZ * HZ * 4);
    float* pooled = (float*)alloc(BZ * HZ * 4);
    float* h0buf  = (float*)alloc(BZ * HZ * 4);
    float* accum  = (float*)alloc(2 * 4);
    (void)ws_size; (void)n_in; (void)in_sizes; (void)out_size;

    auto cvt = [&](const float* s, unsigned short* d, int n) {
        int grid = (n + 255) / 256;
        conv_bf_kernel<<<grid, 256, 0, stream>>>(s, d, n);
    };
    // weight conversions (bf16)
    cvt(gru_whh, whh_bf, FZ * HZ * 3 * HZ);
    conv_wx_kernel<<<(FZ * HZ * 16 + 255) / 256, 256, 0, stream>>>(att_wx, wx_bf);
    cvt(mha_wq, wq_bf, HZ * HZ);  cvt(mha_wk, wk_bf, HZ * HZ);
    cvt(mha_wv, wv_bf, HZ * HZ);  cvt(mha_wo, wo_bf, HZ * HZ);
    cvt(ff_w1, ffw1_bf, HZ * DFFZ);  cvt(ff_w2, ffw2_bf, DFFZ * HZ);
    cvt(fa_wq, faq_bf, HZ * HZ);  cvt(fa_wk, fak_bf, HZ * HZ);
    cvt(fa_wv, fav_bf, HZ * HZ);  cvt(out0_w, o0_bf, HZ * HZ);
    zero2_kernel<<<1, 1, 0, stream>>>(accum);

    // 1) GRU recurrence (WMMA, weight-stationary)
    gru_kernel<<<FZ, 256, 0, stream>>>(x, gru_wih, gru_bih, gru_bhh, whh_bf, Hseq_bf);

    // 2) per-feature time-decay attention -> feats tokens 0..63
    time_attn_kernel<<<dim3(FZ, BZ), 256, 0, stream>>>(Hseq_bf, wx_bf, att_wt, att_rate, feats);
    // 3) demographics token -> feats token 64
    demo_proj_kernel<<<(BZ * HZ + 255) / 256, 256, 0, stream>>>(demo, demo_w, demo_b, feats);

    const int M = BZ * NZ;  // 4160
    dim3 g128((HZ + 63) / 64, (M + 63) / 64);
    dim3 g512((DFFZ + 63) / 64, (M + 63) / 64);
    dim3 gRow((HZ + 63) / 64, 1);

    // 4) MHA
    gemm_bf16_kernel<<<g128, 256, 0, stream>>>(feats, HZ, wq_bf, mha_bq, nullptr, Qb, HZ, M, HZ, HZ, 0);
    gemm_bf16_kernel<<<g128, 256, 0, stream>>>(feats, HZ, wk_bf, mha_bk, nullptr, Kb, HZ, M, HZ, HZ, 0);
    gemm_bf16_kernel<<<g128, 256, 0, stream>>>(feats, HZ, wv_bf, mha_bv, nullptr, Vb, HZ, M, HZ, HZ, 0);
    mha_attn_kernel<<<dim3(BZ, NHD), 128, 0, stream>>>(Qb, Kb, Vb, Xh);
    gemm_bf16_kernel<<<g128, 256, 0, stream>>>(Xh, HZ, wo_bf, mha_bo, nullptr, mha_o, HZ, M, HZ, HZ, 0);

    // 5) DeCov (WMMA c^T c per token)
    decov_center_kernel<<<(NZ * HZ + 255) / 256, 256, 0, stream>>>(mha_o, c_bf, cT_bf);
    decov_cov_kernel<<<NZ, 256, 0, stream>>>(c_bf, cT_bf, accum);
    decov_write_kernel<<<1, 1, 0, stream>>>(accum, dout);

    // 6) residual + FFN
    add_kernel<<<(M * HZ + 255) / 256, 256, 0, stream>>>(feats, mha_o, ctx, M * HZ);
    gemm_bf16_kernel<<<g512, 256, 0, stream>>>(ctx, HZ, ffw1_bf, ff_b1, nullptr, ffh, DFFZ, M, DFFZ, HZ, 1);
    gemm_bf16_kernel<<<g128, 256, 0, stream>>>(ffh, DFFZ, ffw2_bf, ff_b2, ctx, ctx2, HZ, M, HZ, DFFZ, 0);

    // 7) final QKV pooling attention
    gemm_bf16_kernel<<<gRow, 256, 0, stream>>>(ctx2 + (size_t)(NZ - 1) * HZ, NZ * HZ, faq_bf, fa_bq,
                                               nullptr, qf, HZ, BZ, HZ, HZ, 0);
    gemm_bf16_kernel<<<g128, 256, 0, stream>>>(ctx2, HZ, fak_bf, fa_bk, nullptr, kf, HZ, M, HZ, HZ, 0);
    gemm_bf16_kernel<<<g128, 256, 0, stream>>>(ctx2, HZ, fav_bf, fa_bv, nullptr, vf, HZ, M, HZ, HZ, 0);
    fa_pool_kernel<<<BZ, 128, 0, stream>>>(qf, kf, vf, pooled);

    // 8) output head
    gemm_bf16_kernel<<<gRow, 256, 0, stream>>>(pooled, HZ, o0_bf, out0_b, nullptr, h0buf, HZ, BZ, HZ, HZ, 1);
    out_final_kernel<<<1, 64, 0, stream>>>(h0buf, out1_w, out1_b, dout);
}
